// MultiHeadAttention_6004364280076
// MI455X (gfx1250) — matrix operand order repaired and checked by reading, not compiled, not run on hardware
//
#include <hip/hip_runtime.h>
#include <cstdint>

typedef __bf16 bf16;
typedef __attribute__((ext_vector_type(16))) __bf16 v16bf;
typedef __attribute__((ext_vector_type(8)))  __bf16 v8bf;
typedef __attribute__((ext_vector_type(8)))  float  v8f;

#define NUM_HEADS 16
#define HEAD_SIZE 64
#define N_EMBD    1024
#define BATCH     4
#define SEQ       2048
#define MROWS     (BATCH * SEQ)   // 8192 rows of (b,t)

// ---------------------------------------------------------------------------
// CDNA5 async global->LDS copy (ASYNCcnt-tracked DMA into LDS), with a
// synchronous VGPR-roundtrip fallback if the builtins are unavailable.
#if defined(__has_builtin)
#  if __has_builtin(__builtin_amdgcn_global_load_async_to_lds_b128) && \
      __has_builtin(__builtin_amdgcn_s_wait_asynccnt)
#    define USE_ASYNC_LDS 1
#  endif
#endif
#ifndef USE_ASYNC_LDS
#  define USE_ASYNC_LDS 0
#endif

#if USE_ASYNC_LDS
typedef int v4i __attribute__((__vector_size__(16)));
typedef __attribute__((address_space(1))) v4i gas_v4i;   // global (__device__) b128
typedef __attribute__((address_space(3))) v4i las_v4i;   // LDS b128
static __device__ __forceinline__ void async_cp16(void* lds_dst, const void* gsrc) {
  __builtin_amdgcn_global_load_async_to_lds_b128(
      (gas_v4i*)(uintptr_t)gsrc,
      (las_v4i*)(uint32_t)(uintptr_t)lds_dst, 0, 0);
}
#  define ASYNC_CP16(d, s) async_cp16((void*)(d), (const void*)(s))
#  define ASYNC_WAIT()     __builtin_amdgcn_s_wait_asynccnt(0)
#else
#  define ASYNC_CP16(d, s) (*(v8bf*)(d) = *(const v8bf*)(s))
#  define ASYNC_WAIT()     ((void)0)
#endif

static __device__ __forceinline__ bf16 f2bf(float f) { return (bf16)f; }

static __device__ __forceinline__ v16bf cmb(v8bf lo, v8bf hi) {
  v16bf r;
#pragma unroll
  for (int i = 0; i < 8; ++i) { r[i] = lo[i]; r[i + 8] = hi[i]; }
  return r;
}

// ---------------------------------------------------------------------------
// fp32 -> bf16 elementwise
__global__ void k_cvt(const float* __restrict__ src, bf16* __restrict__ dst, int n) {
  int i = blockIdx.x * 256 + threadIdx.x;
  if (i < n) dst[i] = f2bf(src[i]);
}

// proj_w [N][K] fp32 -> projWt [K][N] bf16 (so GEMM computes x @ proj_w.T)
__global__ void k_cvt_t(const float* __restrict__ src, bf16* __restrict__ dst,
                        int N, int K) {
  int i = blockIdx.x * 256 + threadIdx.x;
  if (i < N * K) {
    int n = i / K, k = i % K;
    dst[(size_t)k * N + n] = f2bf(src[i]);
  }
}

// ---------------------------------------------------------------------------
// Tiled bf16 WMMA GEMM: Out[M,N] = A[M,K] @ B[K,N] (+bias). 256 thr = 8 waves.
// Tile 128x64, K-step 32, double-buffered LDS (prefetch tile i+1 while
// computing tile i; one barrier per iteration). A tile staged via async
// global->LDS DMA when available.
template <bool PROJ>
__global__ __launch_bounds__(256) void k_gemm(const bf16* __restrict__ A,
                                              const bf16* __restrict__ Bw,
                                              const float* __restrict__ bias,
                                              void* __restrict__ OutP,
                                              int M, int N, int K) {
  __shared__ __align__(16) bf16 sA[2][128][40];   // +8 pad, 16B-aligned rows
  __shared__ __align__(16) bf16 sBt[2][64][40];   // B tile transposed [n][k]

  const int tid  = threadIdx.x;
  const int lane = tid & 31, w = tid >> 5;
  const int ln   = lane & 15, half = lane >> 4;
  const int m0   = blockIdx.x * 128;
  const int n0   = blockIdx.y * 64;

  const bf16* Bz = Bw;
  bf16*  outb = nullptr;
  float* outf = nullptr;
  if constexpr (PROJ) {
    outf = (float*)OutP;
  } else {
    const int z = blockIdx.z;
    Bz   = Bw + (size_t)z * K * N;
    outb = (bf16*)OutP + (size_t)z * M * N;
  }

  v8f acc[4];
#pragma unroll
  for (int j = 0; j < 4; ++j)
#pragma unroll
    for (int r = 0; r < 8; ++r) acc[j][r] = 0.0f;

  const int rA = tid >> 1, hA = tid & 1;   // A loader: 128 rows x 2 half-rows
  const int kB = tid >> 3, sB = tid & 7;   // B loader: 32 k-rows x 8 segments

  auto load_tile = [&](int k0, int buf) {
    const bf16* ag = A + (size_t)(m0 + rA) * K + k0 + hA * 16;
    ASYNC_CP16(&sA[buf][rA][hA * 16], ag);
    ASYNC_CP16(&sA[buf][rA][hA * 16 + 8], ag + 8);
    const bf16* bg = Bz + (size_t)(k0 + kB) * N + n0 + sB * 8;
    v8bf bt = *(const v8bf*)bg;
#pragma unroll
    for (int e = 0; e < 8; ++e) sBt[buf][sB * 8 + e][kB] = bt[e];
  };

  int buf = 0;
  load_tile(0, 0);
  for (int k0 = 0; k0 < K; k0 += 32) {
    ASYNC_WAIT();
    __syncthreads();                       // tile[buf] ready; prior reads done
    if (k0 + 32 < K) load_tile(k0 + 32, buf ^ 1);

    // A fragment 16x32: lane ln row, K chunks [kk..kk+7] and [kk+16..kk+23]
    v8bf alo = *(const v8bf*)&sA[buf][w * 16 + ln][half * 8];
    v8bf ahi = *(const v8bf*)&sA[buf][w * 16 + ln][half * 8 + 16];
    v16bf a  = cmb(alo, ahi);
#pragma unroll
    for (int j = 0; j < 4; ++j) {
      // B fragment 32x16: lane column n, K chunks [half*8..+7] and [16+half*8..+7]
      const bf16* bp = &sBt[buf][j * 16 + ln][half * 8];
      v16bf b = cmb(*(const v8bf*)bp, *(const v8bf*)(bp + 16));
      acc[j] = __builtin_amdgcn_wmma_f32_16x16x32_bf16(
          false, a, false, b, (short)0, acc[j], false, false);
    }
    buf ^= 1;
  }

  // epilogue: C layout: M = r + 8*half, N = j*16 + ln
#pragma unroll
  for (int j = 0; j < 4; ++j) {
    int col = n0 + j * 16 + ln;
    float badd = 0.0f;
    if constexpr (PROJ) badd = bias[col];
#pragma unroll
    for (int r = 0; r < 8; ++r) {
      int row = m0 + w * 16 + r + half * 8;
      float v = acc[j][r];
      if constexpr (PROJ) outf[(size_t)row * N + col] = v + badd;
      else                outb[(size_t)row * N + col] = f2bf(v);
    }
  }
}

// ---------------------------------------------------------------------------
// Flash attention: one block = 64 q-rows of one (b,h); 128 thr = 4 waves,
// each wave owns a 16-row strip. Streams 64-wide K/V tiles up to the causal
// diagonal with online softmax; double-buffered K/V tiles (async K loads).
__global__ __launch_bounds__(128) void k_attn(const bf16* __restrict__ qkv,
                                              bf16* __restrict__ attn_out) {
  __shared__ __align__(16) bf16 sK[2][64][72];     // [s][d] (== B layout of K^T)
  __shared__ __align__(16) bf16 sVt[2][64][72];    // [d][s] (B layout for P@V)
  __shared__ __align__(16) bf16 sP[4][16][72];     // per-wave P round-trip

  const int tid  = threadIdx.x;
  const int lane = tid & 31, w = tid >> 5;         // 4 waves
  const int ln   = lane & 15, half = lane >> 4;
  const int t0   = blockIdx.x * 64;
  const int h    = blockIdx.y;
  const int b    = blockIdx.z;
  const size_t bT = (size_t)b * SEQ;

  const size_t plane = (size_t)MROWS * HEAD_SIZE;  // per-(z) plane
  const bf16* Qg = qkv + (size_t)(0 * NUM_HEADS + h) * plane;
  const bf16* Kg = qkv + (size_t)(1 * NUM_HEADS + h) * plane;
  const bf16* Vg = qkv + (size_t)(2 * NUM_HEADS + h) * plane;

  // Q strip for this wave, kept in registers: two 16x32 A fragments (d 0..63)
  const bf16* qrow = Qg + (bT + t0 + w * 16 + ln) * HEAD_SIZE;
  v16bf qf[2];
  qf[0] = cmb(*(const v8bf*)(qrow + half * 8),      *(const v8bf*)(qrow + half * 8 + 16));
  qf[1] = cmb(*(const v8bf*)(qrow + 32 + half * 8), *(const v8bf*)(qrow + 32 + half * 8 + 16));

  v8f o[4];
  float m_i[8], l_i[8];
#pragma unroll
  for (int j = 0; j < 4; ++j)
#pragma unroll
    for (int r = 0; r < 8; ++r) o[j][r] = 0.0f;
#pragma unroll
  for (int r = 0; r < 8; ++r) { m_i[r] = -__builtin_inff(); l_i[r] = 0.0f; }

  const int lrow = tid & 63, lseg = tid >> 6;      // tile loaders: 64 rows x 2 segs

  auto load_kv = [&](int s0, int buf) {
    const bf16* krow = Kg + (bT + s0 + lrow) * HEAD_SIZE + lseg * 32;
#pragma unroll
    for (int i = 0; i < 32; i += 8)
      ASYNC_CP16(&sK[buf][lrow][lseg * 32 + i], krow + i);
    const bf16* vrow = Vg + (bT + s0 + lrow) * HEAD_SIZE + lseg * 32;
#pragma unroll
    for (int i = 0; i < 32; i += 8) {
      v8bf t = *(const v8bf*)(vrow + i);
#pragma unroll
      for (int e = 0; e < 8; ++e) sVt[buf][lseg * 32 + i + e][lrow] = t[e];
    }
  };

  const int ntiles = blockIdx.x + 1;               // causal: s-tiles 0..t0/64
  int buf = 0;
  load_kv(0, 0);
  for (int st = 0; st < ntiles; ++st) {
    ASYNC_WAIT();
    __syncthreads();                               // tiles[buf] ready
    if (st + 1 < ntiles) load_kv((st + 1) * 64, buf ^ 1);

    const int s0 = st * 64;

    // --- S = Q K^T (16x64 per wave) ---
    v8f c[4];
#pragma unroll
    for (int j = 0; j < 4; ++j)
#pragma unroll
      for (int r = 0; r < 8; ++r) c[j][r] = 0.0f;
#pragma unroll
    for (int ks = 0; ks < 2; ++ks) {
#pragma unroll
      for (int j = 0; j < 4; ++j) {
        const bf16* bp = &sK[buf][j * 16 + ln][ks * 32 + half * 8];
        v16bf bm = cmb(*(const v8bf*)bp, *(const v8bf*)(bp + 16));
        c[j] = __builtin_amdgcn_wmma_f32_16x16x32_bf16(
            false, qf[ks], false, bm, (short)0, c[j], false, false);
      }
    }

    // --- scale + causal mask ---
#pragma unroll
    for (int j = 0; j < 4; ++j) {
      int s = s0 + j * 16 + ln;
#pragma unroll
      for (int r = 0; r < 8; ++r) {
        int t = t0 + w * 16 + r + half * 8;
        float v = c[j][r] * 0.125f;                // 1/sqrt(64)
        c[j][r] = (s <= t) ? v : -__builtin_inff();
      }
    }

    // --- online softmax (row reductions over 16-lane groups) ---
#pragma unroll
    for (int r = 0; r < 8; ++r) {
      float rm = -__builtin_inff();
#pragma unroll
      for (int j = 0; j < 4; ++j) rm = fmaxf(rm, c[j][r]);
#pragma unroll
      for (int off = 8; off >= 1; off >>= 1) rm = fmaxf(rm, __shfl_xor(rm, off, 32));
      float mnew = fmaxf(m_i[r], rm);
      float corr = __expf(m_i[r] - mnew);
      float rs = 0.0f;
#pragma unroll
      for (int j = 0; j < 4; ++j) {
        float p = __expf(c[j][r] - mnew);
        c[j][r] = p;
        rs += p;
      }
#pragma unroll
      for (int off = 8; off >= 1; off >>= 1) rs += __shfl_xor(rs, off, 32);
      l_i[r] = l_i[r] * corr + rs;
#pragma unroll
      for (int j = 0; j < 4; ++j) o[j][r] *= corr;
      m_i[r] = mnew;
    }

    // --- P: C layout -> LDS -> A layout ---
#pragma unroll
    for (int j = 0; j < 4; ++j)
#pragma unroll
      for (int r = 0; r < 8; ++r)
        sP[w][r + half * 8][j * 16 + ln] = f2bf(c[j][r]);
    __syncthreads();

    // --- O += P @ V ---
#pragma unroll
    for (int ks = 0; ks < 2; ++ks) {
      v8bf alo = *(const v8bf*)&sP[w][ln][ks * 32 + half * 8];
      v8bf ahi = *(const v8bf*)&sP[w][ln][ks * 32 + half * 8 + 16];
      v16bf a  = cmb(alo, ahi);
#pragma unroll
      for (int j = 0; j < 4; ++j) {
        const bf16* bp = &sVt[buf][j * 16 + ln][ks * 32 + half * 8];
        v16bf bm = cmb(*(const v8bf*)bp, *(const v8bf*)(bp + 16));
        o[j] = __builtin_amdgcn_wmma_f32_16x16x32_bf16(
            false, a, false, bm, (short)0, o[j], false, false);
      }
    }
    buf ^= 1;
  }

  // --- normalize and store head output into [row][h*64 + d] layout ---
#pragma unroll
  for (int r = 0; r < 8; ++r) {
    float inv = 1.0f / l_i[r];
    int t = t0 + w * 16 + r + half * 8;
#pragma unroll
    for (int j = 0; j < 4; ++j) {
      attn_out[(bT + t) * N_EMBD + h * HEAD_SIZE + j * 16 + ln] =
          f2bf(o[j][r] * inv);
    }
  }
}

// ---------------------------------------------------------------------------
extern "C" void kernel_launch(void* const* d_in, const int* in_sizes, int n_in,
                              void* d_out, int out_size, void* d_ws, size_t ws_size,
                              hipStream_t stream) {
  const float* x  = (const float*)d_in[0];
  const float* Wk = (const float*)d_in[1];
  const float* Wq = (const float*)d_in[2];
  const float* Wv = (const float*)d_in[3];
  const float* pw = (const float*)d_in[4];
  const float* pb = (const float*)d_in[5];

  // workspace layout (bf16 elements), ~88 MB total
  bf16* xb   = (bf16*)d_ws;                                        // [8192][1024]
  bf16* wqkv = xb   + (size_t)MROWS * N_EMBD;                      // [48][1024][64]
  bf16* pwt  = wqkv + (size_t)3 * NUM_HEADS * N_EMBD * HEAD_SIZE;  // [1024][1024]
  bf16* qkv  = pwt  + (size_t)N_EMBD * N_EMBD;                     // [48][8192][64]
  bf16* attn = qkv  + (size_t)3 * NUM_HEADS * MROWS * HEAD_SIZE;   // [8192][1024]

  const int nx = MROWS * N_EMBD;                  // 8,388,608
  const int nw = NUM_HEADS * N_EMBD * HEAD_SIZE;  // 1,048,576
  const int np = N_EMBD * N_EMBD;                 // 1,048,576

  k_cvt<<<(nx + 255) / 256, 256, 0, stream>>>(x, xb, nx);
  k_cvt<<<(nw + 255) / 256, 256, 0, stream>>>(Wq, wqkv + (size_t)0 * nw, nw);
  k_cvt<<<(nw + 255) / 256, 256, 0, stream>>>(Wk, wqkv + (size_t)1 * nw, nw);
  k_cvt<<<(nw + 255) / 256, 256, 0, stream>>>(Wv, wqkv + (size_t)2 * nw, nw);
  k_cvt_t<<<(np + 255) / 256, 256, 0, stream>>>(pw, pwt, N_EMBD, N_EMBD);

  // Q,K,V projections: batched over 48 (which,head) planes
  k_gemm<false><<<dim3(MROWS / 128, 1, 3 * NUM_HEADS), 256, 0, stream>>>(
      xb, wqkv, nullptr, qkv, MROWS, HEAD_SIZE, N_EMBD);

  // causal flash attention
  k_attn<<<dim3(SEQ / 64, NUM_HEADS, BATCH), 128, 0, stream>>>(qkv, attn);

  // output projection (x @ proj_w.T + b), fp32 out
  k_gemm<true><<<dim3(MROWS / 128, N_EMBD / 64, 1), 256, 0, stream>>>(
      attn, pwt, pb, (float*)d_out, MROWS, N_EMBD, N_EMBD);
}
